// Attention_1425929142307
// MI455X (gfx1250) — matrix operand order repaired and checked by reading, not compiled, not run on hardware
//
#include <hip/hip_runtime.h>
#include <hip/hip_bf16.h>

// ---------- problem constants ----------
static constexpr int Bc  = 2;
static constexpr int Tc  = 2048;
static constexpr int Dc  = 4096;
static constexpr int Hc  = 32;
static constexpr int KVHc = 8;
static constexpr int HDc = 128;          // head dim
static constexpr int NREP = Hc / KVHc;   // 4
static constexpr int Mrows = Bc * Tc;    // 4096 (GEMM M)
#define ATTN_SCALE 0.08838834764831845f  // 1/sqrt(128)
#define LOG2E      1.4426950408889634f

typedef __bf16 bf16;
typedef __attribute__((ext_vector_type(16))) __bf16 v16bf;
typedef __attribute__((ext_vector_type(8)))  __bf16 v8bf;
typedef __attribute__((ext_vector_type(8)))  float  v8f;

// ---------- fragment loaders (contiguous k order, same for A and B) ----------
// A 16x32 bf16: lane holds row M=lane%16; element e -> K = kb + 16g + e (contiguous 16)
__device__ __forceinline__ v16bf load_a_frag(const bf16* __restrict__ row, int kb, int g) {
    return *reinterpret_cast<const v16bf*>(row + kb + 16 * g);
}
// B 32x16 bf16: lane holds col N=lane%16; element e -> K = kb + 16g + e (contiguous 16)
__device__ __forceinline__ v16bf load_b_frag(const bf16* __restrict__ row, int kb, int g) {
    return *reinterpret_cast<const v16bf*>(row + kb + 16 * g);
}

#define WMMA_BF16(A, B, C) \
    __builtin_amdgcn_wmma_f32_16x16x32_bf16(false, (A), false, (B), (short)0, (C), false, false)

// ---------- fp32 -> bf16 convert (vectorized) ----------
__global__ void cvt_bf16_kernel(const float* __restrict__ in, bf16* __restrict__ out, long n4) {
    long i = (long)blockIdx.x * blockDim.x + threadIdx.x;
    if (i >= n4) return;
    const float4 v = reinterpret_cast<const float4*>(in)[i];
    bf16 o[4] = { (bf16)v.x, (bf16)v.y, (bf16)v.z, (bf16)v.w };
    reinterpret_cast<ulong1*>(out)[i] = *reinterpret_cast<const ulong1*>(o);
}

// ---------- GEMM: C[M,N](f32) = A[M,K](bf16,K-major) * B[N,K](bf16,K-major)^T ----------
// each wave computes 32(M) x 64(N); K-loop is ping-pong double buffered (unroll 2).
// launch_bounds(256,2): 8 waves/WGP, <=512 VGPRs/wave -> no spills for ~180 live regs.
__global__ void __launch_bounds__(256, 2)
gemm_bf16_kernel(const bf16* __restrict__ A, const bf16* __restrict__ Bw,
                 float* __restrict__ Cf, int M, int N, int K) {
    const int lane = threadIdx.x & 31;
    const int gwave = blockIdx.x * (blockDim.x >> 5) + (threadIdx.x >> 5);
    const int tilesN = N >> 6;
    const int tilesM = M >> 5;
    if (gwave >= tilesM * tilesN) return;
    const int tm = (gwave / tilesN) * 32;
    const int tn = (gwave % tilesN) * 64;
    const int g = lane >> 4, l16 = lane & 15;

    const bf16* a0  = A  + (size_t)(tm + l16) * K;
    const bf16* a1  = a0 + (size_t)16 * K;
    const bf16* br0 = Bw + (size_t)(tn +  0 + l16) * K;
    const bf16* br1 = Bw + (size_t)(tn + 16 + l16) * K;
    const bf16* br2 = Bw + (size_t)(tn + 32 + l16) * K;
    const bf16* br3 = Bw + (size_t)(tn + 48 + l16) * K;

    v8f acc[2][4] = {};
    // ping-pong fragment buffers
    v16bf pa0, pa1, pb0, pb1, pb2, pb3;   // buffer 0
    v16bf qa0, qa1, qb0, qb1, qb2, qb3;   // buffer 1

    // preload buffer 0 @ k=0
    pa0 = load_a_frag(a0, 0, g);  pa1 = load_a_frag(a1, 0, g);
    pb0 = load_b_frag(br0, 0, g); pb1 = load_b_frag(br1, 0, g);
    pb2 = load_b_frag(br2, 0, g); pb3 = load_b_frag(br3, 0, g);

    int kb = 0;
    for (; kb + 64 < K; kb += 64) {
        // prefetch buffer 1 @ kb+32, then compute buffer 0 @ kb
        qa0 = load_a_frag(a0, kb + 32, g);  qa1 = load_a_frag(a1, kb + 32, g);
        qb0 = load_b_frag(br0, kb + 32, g); qb1 = load_b_frag(br1, kb + 32, g);
        qb2 = load_b_frag(br2, kb + 32, g); qb3 = load_b_frag(br3, kb + 32, g);
        acc[0][0] = WMMA_BF16(pa0, pb0, acc[0][0]);
        acc[1][0] = WMMA_BF16(pa1, pb0, acc[1][0]);
        acc[0][1] = WMMA_BF16(pa0, pb1, acc[0][1]);
        acc[1][1] = WMMA_BF16(pa1, pb1, acc[1][1]);
        acc[0][2] = WMMA_BF16(pa0, pb2, acc[0][2]);
        acc[1][2] = WMMA_BF16(pa1, pb2, acc[1][2]);
        acc[0][3] = WMMA_BF16(pa0, pb3, acc[0][3]);
        acc[1][3] = WMMA_BF16(pa1, pb3, acc[1][3]);
        // prefetch buffer 0 @ kb+64, then compute buffer 1 @ kb+32
        pa0 = load_a_frag(a0, kb + 64, g);  pa1 = load_a_frag(a1, kb + 64, g);
        pb0 = load_b_frag(br0, kb + 64, g); pb1 = load_b_frag(br1, kb + 64, g);
        pb2 = load_b_frag(br2, kb + 64, g); pb3 = load_b_frag(br3, kb + 64, g);
        acc[0][0] = WMMA_BF16(qa0, qb0, acc[0][0]);
        acc[1][0] = WMMA_BF16(qa1, qb0, acc[1][0]);
        acc[0][1] = WMMA_BF16(qa0, qb1, acc[0][1]);
        acc[1][1] = WMMA_BF16(qa1, qb1, acc[1][1]);
        acc[0][2] = WMMA_BF16(qa0, qb2, acc[0][2]);
        acc[1][2] = WMMA_BF16(qa1, qb2, acc[1][2]);
        acc[0][3] = WMMA_BF16(qa0, qb3, acc[0][3]);
        acc[1][3] = WMMA_BF16(qa1, qb3, acc[1][3]);
    }
    // tail: kb == K-64 ; compute kb (buffer0) and kb+32 (buffer1)
    qa0 = load_a_frag(a0, kb + 32, g);  qa1 = load_a_frag(a1, kb + 32, g);
    qb0 = load_b_frag(br0, kb + 32, g); qb1 = load_b_frag(br1, kb + 32, g);
    qb2 = load_b_frag(br2, kb + 32, g); qb3 = load_b_frag(br3, kb + 32, g);
    acc[0][0] = WMMA_BF16(pa0, pb0, acc[0][0]);
    acc[1][0] = WMMA_BF16(pa1, pb0, acc[1][0]);
    acc[0][1] = WMMA_BF16(pa0, pb1, acc[0][1]);
    acc[1][1] = WMMA_BF16(pa1, pb1, acc[1][1]);
    acc[0][2] = WMMA_BF16(pa0, pb2, acc[0][2]);
    acc[1][2] = WMMA_BF16(pa1, pb2, acc[1][2]);
    acc[0][3] = WMMA_BF16(pa0, pb3, acc[0][3]);
    acc[1][3] = WMMA_BF16(pa1, pb3, acc[1][3]);
    acc[0][0] = WMMA_BF16(qa0, qb0, acc[0][0]);
    acc[1][0] = WMMA_BF16(qa1, qb0, acc[1][0]);
    acc[0][1] = WMMA_BF16(qa0, qb1, acc[0][1]);
    acc[1][1] = WMMA_BF16(qa1, qb1, acc[1][1]);
    acc[0][2] = WMMA_BF16(qa0, qb2, acc[0][2]);
    acc[1][2] = WMMA_BF16(qa1, qb2, acc[1][2]);
    acc[0][3] = WMMA_BF16(qa0, qb3, acc[0][3]);
    acc[1][3] = WMMA_BF16(qa1, qb3, acc[1][3]);

    // D layout: lane L, reg r -> row = r + 8g, col = L%16
    #pragma unroll
    for (int i = 0; i < 2; ++i)
        #pragma unroll
        for (int j = 0; j < 4; ++j) {
            const int rbase = tm + 16 * i + 8 * g;
            const int col = tn + 16 * j + l16;
            #pragma unroll
            for (int r = 0; r < 8; ++r)
                Cf[(size_t)(rbase + r) * N + col] = acc[i][j][r];
        }
}

// ---------- RoPE + relayout: f32 [B*T, NH*HD] -> bf16 [B][NH][T][HD], optional prescale ----------
__global__ void rope_relayout_kernel(const float* __restrict__ qf,
                                     const float* __restrict__ fcos,
                                     const float* __restrict__ fsin,
                                     bf16* __restrict__ outBhTd, int NH, float outScale, long total) {
    long idx = (long)blockIdx.x * blockDim.x + threadIdx.x;  // over B*T*NH*(HD/2)
    if (idx >= total) return;
    const int hp = HDc / 2;
    int i = (int)(idx % hp);          idx /= hp;
    int h = (int)(idx % NH);          idx /= NH;
    int t = (int)(idx % Tc);
    int b = (int)(idx / Tc);
    const size_t src = ((size_t)(b * Tc + t) * NH + h) * HDc + 2 * i;
    const float e = qf[src], o = qf[src + 1];
    const float c = fcos[(size_t)t * hp + i], s = fsin[(size_t)t * hp + i];
    const size_t dst = (((size_t)b * NH + h) * Tc + t) * HDc + 2 * i;
    outBhTd[dst]     = (bf16)((e * c - o * s) * outScale);
    outBhTd[dst + 1] = (bf16)((e * s + o * c) * outScale);
}

// ---------- V transpose: f32 [B*T, KVH*HD] -> bf16 Vt[B][KVH][HD][T] ----------
__global__ void v_transpose_kernel(const float* __restrict__ vf, bf16* __restrict__ vt, long total) {
    long idx = (long)blockIdx.x * blockDim.x + threadIdx.x;  // t fastest -> coalesced writes
    if (idx >= total) return;
    int t  = (int)(idx % Tc);          idx /= Tc;
    int hd = (int)(idx % HDc);         idx /= HDc;
    int kv = (int)(idx % KVHc);
    int b  = (int)(idx / KVHc);
    const float v = vf[((size_t)(b * Tc + t) * KVHc + kv) * HDc + hd];
    vt[(((size_t)b * KVHc + kv) * HDc + hd) * Tc + t] = (bf16)v;
}

// ---------- flash attention: one wave = 16 query rows x HD=128, streams keys ----------
// Q is pre-scaled by (1/sqrt(HD))*log2(e); softmax runs in base-2 (native v_exp_f32).
__global__ void __launch_bounds__(256, 2)
flash_attn_kernel(const bf16* __restrict__ Qh,  // [B][H][T][HD]
                  const bf16* __restrict__ Kh,  // [B][KVH][T][HD]
                  const bf16* __restrict__ Vt,  // [B][KVH][HD][T]
                  bf16* __restrict__ ctx) {     // [B*T][H*HD]
    __shared__ __align__(32) bf16 pbuf[8][16][32];   // per-wave 16x32 P tile
    const int lane = threadIdx.x & 31;
    const int wid  = threadIdx.x >> 5;
    const int gwave = blockIdx.x * (blockDim.x >> 5) + wid;
    const int qTiles = Tc / 16;                       // 128
    if (gwave >= Bc * Hc * qTiles) return;
    int r0 = gwave;
    const int qt = r0 % qTiles;  r0 /= qTiles;
    const int h  = r0 % Hc;
    const int b  = r0 / Hc;
    const int kv = h / NREP;
    const int g = lane >> 4, l16 = lane & 15;

    const bf16* qrow  = Qh + (((size_t)b * Hc + h) * Tc + qt * 16 + l16) * HDc;
    const bf16* kbase = Kh + ((size_t)b * KVHc + kv) * Tc * HDc;
    const bf16* vbase = Vt + ((size_t)b * KVHc + kv) * HDc * (size_t)Tc;

    // Q fragments are invariant over the key loop: load once (4 x 32-deep K slices)
    v16bf qa[4];
    #pragma unroll
    for (int kk = 0; kk < 4; ++kk) qa[kk] = load_a_frag(qrow, kk * 32, g);

    float mrun[8], lrun[8];
    #pragma unroll
    for (int r = 0; r < 8; ++r) { mrun[r] = -INFINITY; lrun[r] = 0.f; }
    v8f cacc[8] = {};

    for (int s0 = 0; s0 < Tc; s0 += 32) {
        // ---- S = Q * K^T for a 16x32 key chunk (two 16x16 D tiles) ----
        const bf16* k0 = kbase + (size_t)(s0 +      l16) * HDc;
        const bf16* k1 = kbase + (size_t)(s0 + 16 + l16) * HDc;
        v8f sc0 = {}, sc1 = {};
        #pragma unroll
        for (int kk = 0; kk < 4; ++kk) {
            const int kb = kk * 32;
            v16bf b0 = load_b_frag(k0, kb, g);
            v16bf b1 = load_b_frag(k1, kb, g);
            sc0 = WMMA_BF16(qa[kk], b0, sc0);
            sc1 = WMMA_BF16(qa[kk], b1, sc1);
        }
        // warm L2/L0 for the next chunk while softmax VALU runs
        if (s0 + 32 < Tc) {
            __builtin_prefetch(kbase + (size_t)(s0 + 32 + l16) * HDc, 0, 1);
            __builtin_prefetch(kbase + (size_t)(s0 + 48 + l16) * HDc, 0, 1);
            #pragma unroll
            for (int hdT = 0; hdT < 8; ++hdT)
                __builtin_prefetch(vbase + (size_t)(hdT * 16 + l16) * Tc + s0 + 32, 0, 1);
        }
        // ---- online softmax (base-2) over the 32 new keys; row r+8g lives in one 16-lane half ----
        #pragma unroll
        for (int r = 0; r < 8; ++r) {
            float a = sc0[r], bb = sc1[r];
            float mx = fmaxf(a, bb);
            mx = fmaxf(mx, __shfl_xor(mx, 1, 16));
            mx = fmaxf(mx, __shfl_xor(mx, 2, 16));
            mx = fmaxf(mx, __shfl_xor(mx, 4, 16));
            mx = fmaxf(mx, __shfl_xor(mx, 8, 16));
            const float mnew = fmaxf(mrun[r], mx);
            const float p0 = exp2f(a - mnew), p1 = exp2f(bb - mnew);
            float sum = p0 + p1;
            sum += __shfl_xor(sum, 1, 16);
            sum += __shfl_xor(sum, 2, 16);
            sum += __shfl_xor(sum, 4, 16);
            sum += __shfl_xor(sum, 8, 16);
            const float corr = exp2f(mrun[r] - mnew);
            lrun[r] = lrun[r] * corr + sum;
            mrun[r] = mnew;
            #pragma unroll
            for (int t = 0; t < 8; ++t) cacc[t][r] *= corr;
            pbuf[wid][r + 8 * g][l16]      = (bf16)p0;   // D-layout -> LDS
            pbuf[wid][r + 8 * g][16 + l16] = (bf16)p1;
        }
        // ---- ctx += P(16x32) * V(32x128): P re-read in A-layout from LDS ----
        v16bf pa = load_a_frag(&pbuf[wid][l16][0], 0, g);
        #pragma unroll
        for (int hdT = 0; hdT < 8; ++hdT) {
            const bf16* vrow = vbase + (size_t)(hdT * 16 + l16) * Tc;
            v16bf bv = load_b_frag(vrow, s0, g);
            cacc[hdT] = WMMA_BF16(pa, bv, cacc[hdT]);
        }
    }
    // ---- finalize: divide by l, store bf16 ctx[B*T][H*HD] ----
    const size_t obase = ((size_t)b * Tc + qt * 16) * (size_t)(Hc * HDc) + (size_t)h * HDc;
    #pragma unroll
    for (int r = 0; r < 8; ++r) {
        const float inv = 1.f / lrun[r];
        const size_t rowoff = obase + (size_t)(r + 8 * g) * (Hc * HDc);
        #pragma unroll
        for (int hdT = 0; hdT < 8; ++hdT)
            ctx[rowoff + hdT * 16 + l16] = (bf16)(cacc[hdT][r] * inv);
    }
}

// ---------- host side ----------
static inline void* carve(char*& p, size_t bytes) {
    void* r = (void*)p;
    p += (bytes + 255) & ~(size_t)255;
    return r;
}

extern "C" void kernel_launch(void* const* d_in, const int* in_sizes, int n_in,
                              void* d_out, int out_size, void* d_ws, size_t ws_size,
                              hipStream_t stream) {
    const float* x  = (const float*)d_in[0];
    const float* wq = (const float*)d_in[1];
    const float* wk = (const float*)d_in[2];
    const float* wv = (const float*)d_in[3];
    const float* wo = (const float*)d_in[4];
    const float* fc = (const float*)d_in[5];
    const float* fs = (const float*)d_in[6];
    float* out = (float*)d_out;
    (void)in_sizes; (void)n_in; (void)out_size; (void)ws_size;

    char* p = (char*)d_ws;
    bf16*  xb   = (bf16*) carve(p, (size_t)Mrows * Dc * sizeof(bf16));
    bf16*  wqb  = (bf16*) carve(p, (size_t)Dc * Dc * sizeof(bf16));
    bf16*  wkb  = (bf16*) carve(p, (size_t)(KVHc * HDc) * Dc * sizeof(bf16));
    bf16*  wvb  = (bf16*) carve(p, (size_t)(KVHc * HDc) * Dc * sizeof(bf16));
    bf16*  wob  = (bf16*) carve(p, (size_t)Dc * Dc * sizeof(bf16));
    float* Qf   = (float*)carve(p, (size_t)Mrows * Dc * sizeof(float));
    float* Kf   = (float*)carve(p, (size_t)Mrows * (KVHc * HDc) * sizeof(float));
    float* Vf   = (float*)carve(p, (size_t)Mrows * (KVHc * HDc) * sizeof(float));
    bf16*  Qh   = (bf16*) carve(p, (size_t)Bc * Hc * Tc * HDc * sizeof(bf16));
    bf16*  Khb  = (bf16*) carve(p, (size_t)Bc * KVHc * Tc * HDc * sizeof(bf16));
    bf16*  Vtb  = (bf16*) carve(p, (size_t)Bc * KVHc * HDc * Tc * sizeof(bf16));
    bf16*  ctxb = (bf16*) carve(p, (size_t)Mrows * Dc * sizeof(bf16));

    const int TB = 256;
    auto cvt = [&](const float* src, bf16* dst, long n) {
        long n4 = n / 4;
        cvt_bf16_kernel<<<(int)((n4 + TB - 1) / TB), TB, 0, stream>>>(src, dst, n4);
    };
    // 1) convert inputs to bf16
    cvt(x,  xb,  (long)Mrows * Dc);
    cvt(wq, wqb, (long)Dc * Dc);
    cvt(wk, wkb, (long)(KVHc * HDc) * Dc);
    cvt(wv, wvb, (long)(KVHc * HDc) * Dc);
    cvt(wo, wob, (long)Dc * Dc);

    // 2) Q/K/V projections (WMMA GEMMs, f32 out)
    auto gemm = [&](const bf16* A, const bf16* Bw, float* Cf, int M, int N, int K) {
        int waves = (M >> 5) * (N >> 6);
        int blocks = (waves + 7) / 8;           // 8 waves / 256-thread block
        gemm_bf16_kernel<<<blocks, TB, 0, stream>>>(A, Bw, Cf, M, N, K);
    };
    gemm(xb, wqb, Qf, Mrows, Dc,         Dc);
    gemm(xb, wkb, Kf, Mrows, KVHc * HDc, Dc);
    gemm(xb, wvb, Vf, Mrows, KVHc * HDc, Dc);

    // 3) RoPE + relayout; Q pre-scaled by 1/sqrt(HD)*log2(e) for base-2 softmax
    {
        long tq = (long)Bc * Tc * Hc * (HDc / 2);
        rope_relayout_kernel<<<(int)((tq + TB - 1) / TB), TB, 0, stream>>>(
            Qf, fc, fs, Qh, Hc, ATTN_SCALE * LOG2E, tq);
        long tk = (long)Bc * Tc * KVHc * (HDc / 2);
        rope_relayout_kernel<<<(int)((tk + TB - 1) / TB), TB, 0, stream>>>(
            Kf, fc, fs, Khb, KVHc, 1.0f, tk);
        long tv = (long)Bc * KVHc * HDc * Tc;
        v_transpose_kernel<<<(int)((tv + TB - 1) / TB), TB, 0, stream>>>(Vf, Vtb, tv);
    }

    // 4) flash attention (WMMA for QK^T and PV, streaming base-2 softmax)
    {
        int waves = Bc * Hc * (Tc / 16);        // 8192
        flash_attn_kernel<<<waves / 8, TB, 0, stream>>>(Qh, Khb, Vtb, ctxb);
    }

    // 5) output projection into d_out (f32)
    gemm(ctxb, wob, out, Mrows, Dc, Dc);
}
